// HQQLinear_20504173871125
// MI455X (gfx1250) — matrix operand order repaired and checked by reading, not compiled, not run on hardware
//
#include <hip/hip_runtime.h>
#include <cstdint>
#include <cstddef>

// Problem constants (match reference)
#define IN_F    4096
#define OUT_F   4096
#define GROUP   64
#define GPR     (IN_F / GROUP)   // groups per output row = 64
#define WPR     (IN_F / 8)       // packed "bytes" (stored as int32) per row = 512

// Tiling: 128x128 output tile per 256-thread (8-wave) workgroup.
// Wave grid 4(M) x 2(N); each wave owns 32x64 = 2x4 WMMA tiles.
#define TILE_M  128
#define TILE_N  128
#define TILE_K  64
#define LDSS    72               // LDS row stride in halves (144B -> conflict-free)

typedef _Float16 h16 __attribute__((ext_vector_type(16)));
typedef _Float16 h4  __attribute__((ext_vector_type(4)));
typedef float    f8  __attribute__((ext_vector_type(8)));
typedef float    f4  __attribute__((ext_vector_type(4)));
typedef int      i4  __attribute__((ext_vector_type(4)));

union Frag { h16 v; h4 c[4]; };

__global__ __launch_bounds__(256)
void hqq1bit_wmma(const float* __restrict__ x,
                  const int*   __restrict__ Wp,
                  const float* __restrict__ scale,
                  const float* __restrict__ zero,
                  const float* __restrict__ bias,
                  float* __restrict__ out,
                  int M)
{
    __shared__ __align__(16) _Float16 sA[TILE_M * LDSS];  // x tile, f16
    __shared__ __align__(16) _Float16 sB[TILE_N * LDSS];  // dequantized W tile, f16

    const int tid     = threadIdx.x;
    const int lane    = tid & 31;           // wave32
    const int wave    = tid >> 5;           // 0..7
    const int waveM   = wave >> 1;          // 0..3 -> 32-row slab of the M tile
    const int waveN   = wave & 1;           // 0..1 -> 64-col slab of the N tile
    const int laneLow = lane & 15;
    const int laneHi  = lane >> 4;

    const int nBase = blockIdx.x * TILE_N;
    const int mBase = blockIdx.y * TILE_M;

    f8 acc[2][4] = {};   // 2x4 WMMA tiles per wave, f32 accumulators

    for (int kb = 0; kb < IN_F; kb += TILE_K) {
        // ---------- stage x tile: 128 x 64 f32 -> f16 in LDS ----------
        // 128*64 floats = 2048 float4 slots, 8 per thread (coalesced b128 loads)
        #pragma unroll
        for (int i = 0; i < 8; ++i) {
            int idx = i * 256 + tid;              // 0..2047
            int row = idx >> 4;                   // 16 float4 per row
            int c4  = idx & 15;
            const f4* src = (const f4*)(x + (size_t)(mBase + row) * IN_F + kb + c4 * 4);
            f4 v = *src;
            h4 hv;
            hv.x = (_Float16)v.x; hv.y = (_Float16)v.y;
            hv.z = (_Float16)v.z; hv.w = (_Float16)v.w;
            *(h4*)(&sA[row * LDSS + c4 * 4]) = hv;
        }

        // gfx1250 speculative prefetch of next x tile (no counter cost)
        if (kb + TILE_K < IN_F) {
            __builtin_prefetch(x + (size_t)(mBase + (tid >> 1)) * IN_F
                                 + kb + TILE_K + (tid & 1) * 32, 0, 0);
        }

        // ---------- dequantize W tile: 128 rows x 64 bits -> f16 in LDS ----------
        // Each thread expands 32 bits (4 packed "bytes", one b128 load).
        {
            int r      = tid >> 1;                // 0..127 : output-feature row
            int kLocal = (tid & 1) << 5;          // 0 or 32
            int n      = nBase + r;
            int g      = n * GPR + (kb >> 6);     // one quant group per K-step per row
            float s = scale[g];
            float z = zero[g];
            const i4* wrow = (const i4*)(Wp + (size_t)n * WPR + ((kb + kLocal) >> 3));
            i4 w = *wrow;                         // 4 bytes -> 32 bits, bit 0 first
            #pragma unroll
            for (int byte = 0; byte < 4; ++byte) {
                int bw = w[byte];
                h4 lo, hi;
                #pragma unroll
                for (int b = 0; b < 4; ++b) {
                    lo[b] = (_Float16)(((float)((bw >>  b     ) & 1) - z) * s);
                    hi[b] = (_Float16)(((float)((bw >> (b + 4)) & 1) - z) * s);
                }
                _Float16* dst = &sB[r * LDSS + kLocal + byte * 8];
                ((h4*)dst)[0] = lo;
                ((h4*)dst)[1] = hi;
            }
        }
        __syncthreads();

        // ---------- two K=32 WMMA chunks, 8 WMMAs each ----------
        #pragma unroll
        for (int kc = 0; kc < 2; ++kc) {
            // A fragments: 16x32 f16.  lane<16: K 0-15 ; lane>=16: K 16-31
            Frag a[2];
            #pragma unroll
            for (int i = 0; i < 2; ++i) {
                const _Float16* base =
                    &sA[(waveM * 32 + i * 16 + laneLow) * LDSS + kc * 32];
                int kh = laneHi * 16;
                a[i].c[0] = *(const h4*)(base + kh);
                a[i].c[1] = *(const h4*)(base + kh + 4);
                a[i].c[2] = *(const h4*)(base + kh + 8);
                a[i].c[3] = *(const h4*)(base + kh + 12);
            }
            // B fragments: 32x16 f16, B[k][n] = What[n][k].
            // lane%16 = column n, lane/16 selects K half (0-15 / 16-31)
            Frag bf[4];
            #pragma unroll
            for (int j = 0; j < 4; ++j) {
                const _Float16* base =
                    &sB[(waveN * 64 + j * 16 + laneLow) * LDSS + kc * 32 + laneHi * 16];
                bf[j].c[0] = *(const h4*)(base);
                bf[j].c[1] = *(const h4*)(base + 4);
                bf[j].c[2] = *(const h4*)(base + 8);
                bf[j].c[3] = *(const h4*)(base + 12);
            }
            #pragma unroll
            for (int i = 0; i < 2; ++i)
                #pragma unroll
                for (int j = 0; j < 4; ++j)
                    acc[i][j] = __builtin_amdgcn_wmma_f32_16x16x32_f16(
                        /*neg_a=*/false, a[i].v,
                        /*neg_b=*/false, bf[j].v,
                        /*c_mod=*/(short)0, acc[i][j],
                        /*reuse_a=*/false, /*reuse_b=*/false);
        }
        __syncthreads();
    }

    // ---------- epilogue: C/D layout -> global, fused bias ----------
    // VGPR v: lanes 0-15 -> M=v, N=lane ; lanes 16-31 -> M=8+v, N=lane-16
    #pragma unroll
    for (int j = 0; j < 4; ++j) {
        int nCol = nBase + waveN * 64 + j * 16 + laneLow;
        float bv = bias[nCol];
        #pragma unroll
        for (int i = 0; i < 2; ++i) {
            int m0 = mBase + waveM * 32 + i * 16 + laneHi * 8;
            #pragma unroll
            for (int v = 0; v < 8; ++v) {
                out[(size_t)(m0 + v) * OUT_F + nCol] = acc[i][j][v] + bv;
            }
        }
    }
}

extern "C" void kernel_launch(void* const* d_in, const int* in_sizes, int n_in,
                              void* d_out, int out_size, void* d_ws, size_t ws_size,
                              hipStream_t stream) {
    const float* x     = (const float*)d_in[0];
    const int*   Wp    = (const int*)d_in[1];
    const float* scale = (const float*)d_in[2];
    const float* zero  = (const float*)d_in[3];
    const float* bias  = (const float*)d_in[4];
    float*       out   = (float*)d_out;

    const int M = in_sizes[0] / IN_F;   // 8192

    dim3 grid(OUT_F / TILE_N, M / TILE_M);   // (32, 64)
    dim3 block(256);                          // 8 wave32 waves
    hqq1bit_wmma<<<grid, block, 0, stream>>>(x, Wp, scale, zero, bias, out, M);
}
